// STN_25434796327644
// MI455X (gfx1250) — matrix operand order repaired and checked by reading, not compiled, not run on hardware
//
#include <hip/hip_runtime.h>

typedef __attribute__((ext_vector_type(16))) __bf16 v16bf;
typedef __attribute__((ext_vector_type(8)))  __bf16 v8bf;
typedef __attribute__((ext_vector_type(8)))  float  v8f;
typedef int v4i_b128 __attribute__((vector_size(16)));   // matches async-LDS builtin params

#define DIM   3072
#define BATCH 16384
#define BM    128
#define BN    256
#define BK    32
#define LDA   40   // BK + 8 halves padding -> conflict-free b128 LDS reads
#define LDB   40
#define KT    (DIM / BK)   // 96

#if __has_builtin(__builtin_amdgcn_global_load_async_to_lds_b128)
#define HAVE_ASYNC_LDS 1
#else
#define HAVE_ASYNC_LDS 0
#endif

#define AS1 __attribute__((address_space(1)))
#define AS3 __attribute__((address_space(3)))

__device__ __forceinline__ void wait_asynccnt0() {
#if __has_builtin(__builtin_amdgcn_s_wait_asynccnt)
  __builtin_amdgcn_s_wait_asynccnt(0);
#else
  asm volatile("s_wait_asynccnt 0x0" ::: "memory");
#endif
}

// ---------------- pack kernels (fast path) ----------------

__global__ void pack_x_bf16(const float* __restrict__ x, __bf16* __restrict__ xb) {
  size_t i = ((size_t)blockIdx.x * blockDim.x + threadIdx.x) * 8;
  float4 a = *(const float4*)(x + i);
  float4 c = *(const float4*)(x + i + 4);
  v8bf o;
  o[0] = (__bf16)a.x; o[1] = (__bf16)a.y; o[2] = (__bf16)a.z; o[3] = (__bf16)a.w;
  o[4] = (__bf16)c.x; o[5] = (__bf16)c.y; o[6] = (__bf16)c.z; o[7] = (__bf16)c.w;
  *(v8bf*)(xb + i) = o;
}

// w_eff^T[u][d] = mask[u][d] * w[d][u], bf16, d contiguous (K-contiguous for GEMM B)
__global__ void pack_weff(const float* __restrict__ w, const float* __restrict__ mask,
                          __bf16* __restrict__ wt) {
  __shared__ float tile[32][33];
  const int u0 = blockIdx.x * 32;
  const int d0 = blockIdx.y * 32;
  const int tx = threadIdx.x & 31;
  const int ty = threadIdx.x >> 5;          // 0..7
#pragma unroll
  for (int r = 0; r < 32; r += 8)
    tile[tx][ty + r] = w[(size_t)(d0 + ty + r) * DIM + (u0 + tx)];  // tile[u][d] = w[d][u]
  __syncthreads();
#pragma unroll
  for (int r = 0; r < 32; r += 8) {
    int u = u0 + ty + r, d = d0 + tx;
    wt[(size_t)u * DIM + d] = (__bf16)(mask[(size_t)u * DIM + d] * tile[ty + r][tx]);
  }
}

// ---------------- shared GEMM pieces ----------------

// Keep only B fragments (32 VGPRs) live; stream one A fragment per tm.
__device__ __forceinline__ void mma_step(const __bf16* As, const __bf16* Bs,
                                         v8f (&acc)[4][4], int wm, int wn, int lr, int lh) {
  v16bf bfr[4];
#pragma unroll
  for (int tn = 0; tn < 4; tn++) {
    const __bf16* p = &Bs[(wn * 64 + tn * 16 + lr) * LDB + lh * 16];
    v8bf lo = *(const v8bf*)p;            // K = lh*16 .. +7
    v8bf hi = *(const v8bf*)(p + 8);      // K = lh*16+8 .. +15
    bfr[tn] = __builtin_shufflevector(lo, hi, 0,1,2,3,4,5,6,7,8,9,10,11,12,13,14,15);
  }
#pragma unroll
  for (int tm = 0; tm < 4; tm++) {
    const __bf16* p = &As[(wm * 64 + tm * 16 + lr) * LDA + lh * 16];
    v8bf lo = *(const v8bf*)p;            // K = lh*16 .. +7
    v8bf hi = *(const v8bf*)(p + 8);      // K = lh*16+8 .. +15
    v16bf af = __builtin_shufflevector(lo, hi, 0,1,2,3,4,5,6,7,8,9,10,11,12,13,14,15);
#pragma unroll
    for (int tn = 0; tn < 4; tn++)
      acc[tm][tn] = __builtin_amdgcn_wmma_f32_16x16x32_bf16(
          false, af, false, bfr[tn], (short)0, acc[tm][tn], false, false);
  }
}

__device__ __forceinline__ void epilogue(v8f (&acc)[4][4], const float* __restrict__ bias,
                                         float* __restrict__ out,
                                         int m0, int n0, int wm, int wn, int lr, int lh) {
#pragma unroll
  for (int tn = 0; tn < 4; tn++) {
    const int col = n0 + wn * 64 + tn * 16 + lr;
    const float bv = bias[col];
#pragma unroll
    for (int tm = 0; tm < 4; tm++) {
      const int mbase = m0 + wm * 64 + tm * 16 + lh * 8;
      float* op = out + (size_t)mbase * DIM + col;
#pragma unroll
      for (int r = 0; r < 8; r++)
        op[(size_t)r * DIM] = acc[tm][tn][r] + bv;
    }
  }
}

// ---------------- fast path: LDS-double-buffered async-copy GEMM ----------------

__global__ __launch_bounds__(256)
void stn_gemm_bf16(const __bf16* __restrict__ A,    // x_bf16 [BATCH][DIM]
                   const __bf16* __restrict__ Bt,   // w_eff^T [DIM(u)][DIM(d)]
                   const float*  __restrict__ bias,
                   float* __restrict__ out) {
  __shared__ __bf16 As[2][BM * LDA];   // 2 x 10.0 KB
  __shared__ __bf16 Bs[2][BN * LDB];   // 2 x 20.0 KB

  const int tid = threadIdx.x;
  const int lane = tid & 31, wave = tid >> 5;
  const int wm = wave & 1, wn = wave >> 1;
  const int lr = lane & 15, lh = lane >> 4;
  const int m0 = blockIdx.y * BM;
  const int n0 = blockIdx.x * BN;

  // Staging unit: (row = tid>>2, seg = tid&3); unit stride = 64 rows.
  const int srow = tid >> 2, sseg = tid & 3;
  const __bf16* aB = A  + (size_t)(m0 + srow) * DIM + sseg * 8;
  const __bf16* bB = Bt + (size_t)(n0 + srow) * DIM + sseg * 8;
  const int aLo = srow * LDA + sseg * 8;
  const int bLo = srow * LDB + sseg * 8;

  v8f acc[4][4];
  const v8f zero = {0.f, 0.f, 0.f, 0.f, 0.f, 0.f, 0.f, 0.f};
#pragma unroll
  for (int a = 0; a < 4; a++)
#pragma unroll
    for (int b = 0; b < 4; b++) acc[a][b] = zero;

#if HAVE_ASYNC_LDS
  // --- zero-VGPR staging via GLOBAL_LOAD_ASYNC_TO_LDS_B128 (ASYNCcnt) ---
  #define ISSUE_TILE(buf, gA, gB)                                                   \
    do {                                                                            \
      _Pragma("unroll")                                                             \
      for (int i = 0; i < 2; i++)                                                   \
        __builtin_amdgcn_global_load_async_to_lds_b128(                             \
            (AS1 v4i_b128*)(gA + (size_t)i * 64 * DIM),                             \
            (AS3 v4i_b128*)(&As[buf][aLo + i * 64 * LDA]), 0, 0);                   \
      _Pragma("unroll")                                                             \
      for (int i = 0; i < 4; i++)                                                   \
        __builtin_amdgcn_global_load_async_to_lds_b128(                             \
            (AS1 v4i_b128*)(gB + (size_t)i * 64 * DIM),                             \
            (AS3 v4i_b128*)(&Bs[buf][bLo + i * 64 * LDB]), 0, 0);                   \
    } while (0)

  ISSUE_TILE(0, aB, bB);
  wait_asynccnt0();
  __syncthreads();

  for (int t = 0; t < KT; ++t) {
    if (t + 1 < KT) {                  // fill the other buffer while WMMAs run
      ISSUE_TILE((t + 1) & 1, (aB + BK), (bB + BK));
      if (t + 2 < KT) {
        __builtin_prefetch(aB + 2 * BK, 0, 2);
        __builtin_prefetch(bB + 2 * BK, 0, 2);
      }
    }
    mma_step(&As[t & 1][0], &Bs[t & 1][0], acc, wm, wn, lr, lh);
    wait_asynccnt0();                  // next buffer's DMA done (this wave)
    __syncthreads();                   // ... and everyone else's
    aB += BK; bB += BK;
  }
  #undef ISSUE_TILE
#else
  // --- synchronous staging fallback (single buffer) ---
  for (int t = 0; t < KT; ++t) {
    uint4 ar[2], br[4];
#pragma unroll
    for (int i = 0; i < 2; i++) ar[i] = *(const uint4*)(aB + (size_t)i * 64 * DIM);
#pragma unroll
    for (int i = 0; i < 4; i++) br[i] = *(const uint4*)(bB + (size_t)i * 64 * DIM);
#pragma unroll
    for (int i = 0; i < 2; i++) *(uint4*)(&As[0][aLo + i * 64 * LDA]) = ar[i];
#pragma unroll
    for (int i = 0; i < 4; i++) *(uint4*)(&Bs[0][bLo + i * 64 * LDB]) = br[i];
    __syncthreads();
    mma_step(&As[0][0], &Bs[0][0], acc, wm, wn, lr, lh);
    __syncthreads();
    aB += BK; bB += BK;
  }
#endif

  epilogue(acc, bias, out, m0, n0, wm, wn, lr, lh);
}

// ---------------- fallback: fully fused, no workspace ----------------

__global__ __launch_bounds__(256)
void stn_gemm_fused(const float* __restrict__ x, const float* __restrict__ w,
                    const float* __restrict__ mask, const float* __restrict__ bias,
                    float* __restrict__ out) {
  __shared__ __bf16 As[BM * LDA];
  __shared__ __bf16 Bs[BN * LDB];

  const int tid = threadIdx.x;
  const int lane = tid & 31, wave = tid >> 5;
  const int wm = wave & 1, wn = wave >> 1;
  const int lr = lane & 15, lh = lane >> 4;
  const int m0 = blockIdx.y * BM;
  const int n0 = blockIdx.x * BN;

  v8f acc[4][4];
  const v8f zero = {0.f, 0.f, 0.f, 0.f, 0.f, 0.f, 0.f, 0.f};
#pragma unroll
  for (int a = 0; a < 4; a++)
#pragma unroll
    for (int b = 0; b < 4; b++) acc[a][b] = zero;

  for (int t = 0; t < KT; ++t) {
    const int kk = t * BK;
#pragma unroll
    for (int i = 0; i < 4; i++) {     // A: 128 rows x 8 float4 segs
      int u = tid + i * 256; int row = u >> 3, seg = u & 7;
      float4 v = *(const float4*)(x + (size_t)(m0 + row) * DIM + kk + seg * 4);
      union { uint2 q; __bf16 h[4]; } cv;
      cv.h[0] = (__bf16)v.x; cv.h[1] = (__bf16)v.y;
      cv.h[2] = (__bf16)v.z; cv.h[3] = (__bf16)v.w;
      *(uint2*)(&As[row * LDA + seg * 4]) = cv.q;
    }
#pragma unroll
    for (int i = 0; i < 8; i++) {     // B: w_eff on the fly
      int u = tid + i * 256; int row = u >> 3, seg = u & 7;
      int n = n0 + row, kb = kk + seg * 4;
      float4 mv = *(const float4*)(mask + (size_t)n * DIM + kb);
      union { uint2 q; __bf16 h[4]; } cv;
      cv.h[0] = (__bf16)(mv.x * w[(size_t)(kb + 0) * DIM + n]);
      cv.h[1] = (__bf16)(mv.y * w[(size_t)(kb + 1) * DIM + n]);
      cv.h[2] = (__bf16)(mv.z * w[(size_t)(kb + 2) * DIM + n]);
      cv.h[3] = (__bf16)(mv.w * w[(size_t)(kb + 3) * DIM + n]);
      *(uint2*)(&Bs[row * LDB + seg * 4]) = cv.q;
    }
    __syncthreads();
    mma_step(As, Bs, acc, wm, wn, lr, lh);
    __syncthreads();
  }

  epilogue(acc, bias, out, m0, n0, wm, wn, lr, lh);
}

// ---------------- host ----------------

extern "C" void kernel_launch(void* const* d_in, const int* in_sizes, int n_in,
                              void* d_out, int out_size, void* d_ws, size_t ws_size,
                              hipStream_t stream) {
  const float* x    = (const float*)d_in[0];
  const float* w    = (const float*)d_in[1];
  const float* b    = (const float*)d_in[2];
  const float* mask = (const float*)d_in[3];
  float* out = (float*)d_out;

  const size_t XB = (size_t)BATCH * DIM * 2;   // x in bf16
  const size_t WB = (size_t)DIM * DIM * 2;     // w_eff^T in bf16
  dim3 ggrid(DIM / BN, BATCH / BM);            // (12, 128)

  if (ws_size >= XB + WB) {
    __bf16* xb = (__bf16*)d_ws;
    __bf16* wt = (__bf16*)((char*)d_ws + XB);
    const size_t nvec = (size_t)BATCH * DIM / 8;      // 6291456, /256 exact
    pack_x_bf16<<<dim3((unsigned)(nvec / 256)), 256, 0, stream>>>(x, xb);
    pack_weff<<<dim3(DIM / 32, DIM / 32), 256, 0, stream>>>(w, mask, wt);
    stn_gemm_bf16<<<ggrid, 256, 0, stream>>>(xb, wt, b, out);
  } else {
    stn_gemm_fused<<<ggrid, 256, 0, stream>>>(x, w, mask, b, out);
  }
}
